// Discriminator_85676007620769
// MI455X (gfx1250) — compile-verified, matrix operand order checked by reading, not run on hardware
//
#include <hip/hip_runtime.h>
#include <hip/hip_bf16.h>
#include <stdint.h>

// ---------------------------------------------------------------------------
// CDNA5 / gfx1250 implementation (wave32, WMMA f16->f32).
// Pipeline: fused 3x conv -> f16 A ; WMMA GEMMs (FC1,FC2,XGates,Disc1,Disc2)
// with double-buffered async global->LDS tile copies; sequential LSTM with
// 8 cooperating blocks + WMMA recurrent GEMM.
// ---------------------------------------------------------------------------

typedef __attribute__((ext_vector_type(16))) _Float16 v16h;
typedef __attribute__((ext_vector_type(8)))  _Float16 v8h;
typedef __attribute__((ext_vector_type(8)))  float    v8f;

#define TT 64
#define BB 32
#define HID 512
#define ZD 64
#define CH 4
#define HH 84
#define WW 84
#define IMG (HH*WW)           // 7056
#define FEAT_IN (CH*IMG)      // 28224
#define ROWS (TT*BB)          // 2048
#define SLOPE 0.01f

__device__ __forceinline__ float lrelu(float v) { return v > 0.f ? v : SLOPE * v; }
__device__ __forceinline__ float sigmf(float v) { return 1.f / (1.f + __expf(-v)); }

// Build a 16x16x32 f16 A/B fragment for this lane.
// Per ISA 7.12.2 (16-bit A 16x32): lanes 0-15 hold K = koff+[0..7] and koff+16+[0..7],
// lanes 16-31 hold K = koff+8+[0..7] and koff+24+[0..7]; row/col = lane%16.
__device__ __forceinline__ v16h frag16(const _Float16* p, int koff) {
  v8h lo = *(const v8h*)(p + koff);
  v8h hi = *(const v8h*)(p + koff + 16);
  v16h r;
#pragma unroll
  for (int j = 0; j < 8; ++j) { r[j] = lo[j]; r[8 + j] = hi[j]; }
  return r;
}

// Async global->LDS copy of 16 bytes (CDNA5 GLOBAL_LOAD_ASYNC_TO_LDS_B128,
// GV addressing, tracked by ASYNCcnt). ldsaddr = low 32 bits of generic
// shared-memory address (LDS aperture truncation per ISA 10.2).
__device__ __forceinline__ void async_copy_b128(void* lds_dst, const void* gsrc) {
  unsigned int la = (unsigned int)(size_t)lds_dst;
  unsigned long long ga = (unsigned long long)(size_t)gsrc;
  asm volatile("global_load_async_to_lds_b128 %0, %1, off" ::"v"(la), "v"(ga) : "memory");
}

// ---------------------------------------------------------------------------
// Utility kernels
// ---------------------------------------------------------------------------
__global__ void k_cvt_f32_f16(const float* __restrict__ s, _Float16* __restrict__ d, int n) {
  int i = blockIdx.x * blockDim.x + threadIdx.x;
  if (i < n) d[i] = (_Float16)s[i];
}

__global__ void k_zero_u32(unsigned int* __restrict__ p, int n) {
  int i = blockIdx.x * blockDim.x + threadIdx.x;
  if (i < n) p[i] = 0u;
}

// ---------------------------------------------------------------------------
// Fused conv1(5x5) -> conv2(3x3) -> conv3(3x3), all SAME, 4 channels, lrelu.
// One 16x16 output tile per block; halo 4 -> 24x24 input tile in LDS.
// Output written as f16 row-major A matrix [2048, 28224].
// ---------------------------------------------------------------------------
__global__ void k_conv_fused(const float* __restrict__ x,
                             const float* __restrict__ w1, const float* __restrict__ b1,
                             const float* __restrict__ w2, const float* __restrict__ b2,
                             const float* __restrict__ w3, const float* __restrict__ b3,
                             _Float16* __restrict__ A1) {
  __shared__ float in_s[CH][24][24];
  __shared__ float t1[CH][20][20];
  __shared__ float t2[CH][18][18];
  __shared__ float w1s[CH][CH][5][5];
  __shared__ float w2s[CH][CH][3][3];
  __shared__ float w3s[CH][CH][3][3];
  __shared__ float b1s[CH], b2s[CH], b3s[CH];

  const int tid = threadIdx.x;
  const int n = blockIdx.z;
  const int x0 = (int)blockIdx.x * 16 - 4;
  const int y0 = (int)blockIdx.y * 16 - 4;

  for (int i = tid; i < CH * CH * 25; i += 256) ((float*)w1s)[i] = w1[i];
  for (int i = tid; i < CH * CH * 9; i += 256) { ((float*)w2s)[i] = w2[i]; ((float*)w3s)[i] = w3[i]; }
  if (tid < CH) { b1s[tid] = b1[tid]; b2s[tid] = b2[tid]; b3s[tid] = b3[tid]; }

  const float* xn = x + (size_t)n * CH * IMG;
  for (int i = tid; i < CH * 24 * 24; i += 256) {
    int c = i / (24 * 24), r = i % (24 * 24), yy = r / 24, xx = r % 24;
    int gx = x0 + xx, gy = y0 + yy;
    float v = 0.f;
    if (gx >= 0 && gx < WW && gy >= 0 && gy < HH) v = xn[c * IMG + gy * WW + gx];
    in_s[c][yy][xx] = v;
  }
  __syncthreads();

  // conv1: 20x20 region, global = (x0+2+px, y0+2+py)
  for (int i = tid; i < CH * 20 * 20; i += 256) {
    int co = i / 400, r = i % 400, py = r / 20, px = r % 20;
    int gx = x0 + 2 + px, gy = y0 + 2 + py;
    float acc = b1s[co];
#pragma unroll
    for (int ci = 0; ci < CH; ++ci)
#pragma unroll
      for (int ky = 0; ky < 5; ++ky)
#pragma unroll
        for (int kx = 0; kx < 5; ++kx)
          acc = fmaf(w1s[co][ci][ky][kx], in_s[ci][py + ky][px + kx], acc);
    acc = lrelu(acc);
    t1[co][py][px] = (gx >= 0 && gx < WW && gy >= 0 && gy < HH) ? acc : 0.f;
  }
  __syncthreads();

  // conv2: 18x18 region, global = (x0+3+px, y0+3+py)
  for (int i = tid; i < CH * 18 * 18; i += 256) {
    int co = i / 324, r = i % 324, py = r / 18, px = r % 18;
    int gx = x0 + 3 + px, gy = y0 + 3 + py;
    float acc = b2s[co];
#pragma unroll
    for (int ci = 0; ci < CH; ++ci)
#pragma unroll
      for (int ky = 0; ky < 3; ++ky)
#pragma unroll
        for (int kx = 0; kx < 3; ++kx)
          acc = fmaf(w2s[co][ci][ky][kx], t1[ci][py + ky][px + kx], acc);
    acc = lrelu(acc);
    t2[co][py][px] = (gx >= 0 && gx < WW && gy >= 0 && gy < HH) ? acc : 0.f;
  }
  __syncthreads();

  // conv3: 16x16 outputs, global = (blockIdx.x*16+px, blockIdx.y*16+py)
  for (int i = tid; i < CH * 16 * 16; i += 256) {
    int co = i / 256, r = i % 256, py = r / 16, px = r % 16;
    int gx = (int)blockIdx.x * 16 + px, gy = (int)blockIdx.y * 16 + py;
    if (gx >= WW || gy >= HH) continue;
    float acc = b3s[co];
#pragma unroll
    for (int ci = 0; ci < CH; ++ci)
#pragma unroll
      for (int ky = 0; ky < 3; ++ky)
#pragma unroll
        for (int kx = 0; kx < 3; ++kx)
          acc = fmaf(w3s[co][ci][ky][kx], t2[ci][py + ky][px + kx], acc);
    acc = lrelu(acc);
    A1[(size_t)n * FEAT_IN + co * IMG + gy * WW + gx] = (_Float16)acc;
  }
}

// ---------------------------------------------------------------------------
// WMMA GEMM: out[M,N] = act(A[M,K](f16) * W[N,K](f16)^T + bias [+ bias2])
// Block tile 128x64, 8 waves (4 M-slots x 2 N-slots), wave tile 32x32
// (2x2 fragments of v_wmma_f32_16x16x32_f16). K tiled by 32 through LDS,
// double-buffered with async global->LDS copies (ASYNCcnt) so the 23.3 TB/s
// fetch overlaps the matrix pipes; no VGPR staging round-trip.
// ---------------------------------------------------------------------------
__global__ void k_wmma_gemm(const _Float16* __restrict__ A, const _Float16* __restrict__ W,
                            const float* __restrict__ bias, const float* __restrict__ bias2,
                            _Float16* __restrict__ out16, float* __restrict__ out32,
                            int M, int N, int K, int act) {
  __shared__ __align__(16) _Float16 As[2][128][40];  // pad 40 breaks bank conflicts
  __shared__ __align__(16) _Float16 Bs[2][64][40];

  const int tid = threadIdx.x;
  const int lane = tid & 31, wave = tid >> 5;
  const int wm = wave & 3, wn = wave >> 2;
  const int lm = lane & 15, khalf = lane >> 4;
  const int mBlock = (int)blockIdx.x * 128;
  const int nBlock = (int)blockIdx.y * 64;

  const int arow = tid >> 1, aseg = tid & 1;   // A: 16 f16 / thread = 2 x b128
  const int brow = tid >> 2, bseg = tid & 3;   // B: 8 f16 / thread  = 1 x b128

  // Each thread issues 3 async b128 copies per K-tile (ASYNCcnt +3 per wave).
  auto issue_tile = [&](int kc, int buf) {
    const _Float16* ga = A + (size_t)(mBlock + arow) * K + kc + aseg * 16;
    async_copy_b128(&As[buf][arow][aseg * 16], ga);
    async_copy_b128(&As[buf][arow][aseg * 16 + 8], ga + 8);
    const _Float16* gb = W + (size_t)(nBlock + brow) * K + kc + bseg * 8;
    async_copy_b128(&Bs[buf][brow][bseg * 8], gb);
  };

  v8f acc[2][2] = {};

  issue_tile(0, 0);
  int buf = 0;
  for (int kc = 0; kc < K; kc += 32, buf ^= 1) {
    if (kc + 32 < K) {
      issue_tile(kc + 32, buf ^ 1);      // prefetch next tile into other buffer
      if (kc + 64 < K)
        __builtin_prefetch(A + (size_t)(mBlock + arow) * K + kc + 64, 0, 0);
      // retire current tile's 3 copies; leave the 3 just issued in flight
      asm volatile("s_wait_asynccnt 0x3" ::: "memory");
    } else {
      asm volatile("s_wait_asynccnt 0x0" ::: "memory");
    }
    __syncthreads();                     // LDS tile visible to all waves

    v16h aF[2], bF[2];
#pragma unroll
    for (int mi = 0; mi < 2; ++mi) aF[mi] = frag16(&As[buf][wm * 32 + mi * 16 + lm][0], khalf * 8);
#pragma unroll
    for (int ni = 0; ni < 2; ++ni) bF[ni] = frag16(&Bs[buf][wn * 32 + ni * 16 + lm][0], khalf * 8);
#pragma unroll
    for (int mi = 0; mi < 2; ++mi)
#pragma unroll
      for (int ni = 0; ni < 2; ++ni)
        acc[mi][ni] = __builtin_amdgcn_wmma_f32_16x16x32_f16(
            false, aF[mi], false, bF[ni], (short)0, acc[mi][ni], false, false);
    __syncthreads();                     // buffer reusable by the issue 2 tiles ahead
  }

  // Epilogue. C/D layout: lanes 0-15 -> M = r, lanes 16-31 -> M = 8+r ; N = lane%16.
  const int moff = (lane >> 4) * 8;
#pragma unroll
  for (int mi = 0; mi < 2; ++mi) {
#pragma unroll
    for (int ni = 0; ni < 2; ++ni) {
      int n = nBlock + wn * 32 + ni * 16 + lm;
      float bv = (bias ? bias[n] : 0.f) + (bias2 ? bias2[n] : 0.f);
#pragma unroll
      for (int r = 0; r < 8; ++r) {
        int m = mBlock + wm * 32 + mi * 16 + moff + r;
        float v = acc[mi][ni][r] + bv;
        if (act) v = lrelu(v);
        if (out32) out32[(size_t)m * N + n] = v;
        if (out16) out16[(size_t)m * N + n] = (_Float16)v;
      }
    }
  }
}

// ---------------------------------------------------------------------------
// Sequential LSTM. XG[2048,2048] f32 already holds x@W_ih^T + b_ih + b_hh.
// 8 blocks; block k owns hidden cols [k*64,(k+1)*64) -> gate cols g*512+k*64+j.
// ---------------------------------------------------------------------------
#define LSTM_BLOCKS 8

__device__ __forceinline__ void grid_barrier(unsigned int* cnt) {
  __syncthreads();
  if (threadIdx.x == 0) {
    __threadfence();
    (void)__hip_atomic_fetch_add(cnt, 1u, __ATOMIC_ACQ_REL, __HIP_MEMORY_SCOPE_AGENT);
    while (__hip_atomic_load(cnt, __ATOMIC_ACQUIRE, __HIP_MEMORY_SCOPE_AGENT) <
           (unsigned int)LSTM_BLOCKS)
      __builtin_amdgcn_s_sleep(1);
  }
  __syncthreads();
}

__global__ void k_lstm_seq(const float* __restrict__ XG,       // [2048,2048]
                           const _Float16* __restrict__ Whh,   // [2048,512] f16
                           const float* __restrict__ done,     // [2048]
                           const float* __restrict__ c0,       // [32,512]
                           _Float16* __restrict__ hbuf,        // [32,512] f16 (init from h0)
                           _Float16* __restrict__ feat,        // [2048,512] f16 (ys)
                           float* __restrict__ outHN, float* __restrict__ outCN,
                           unsigned int* __restrict__ sync) {
  // hp_s (32 x 520 f16 = 33280 B) overlays gates_s (32 x 256 f32 = 32768 B).
  __shared__ __align__(16) unsigned char smem[33280];
  __shared__ float c_s[BB][64];

  _Float16(*hp_s)[520] = (_Float16(*)[520])smem;
  float(*gates_s)[256] = (float(*)[256])smem;

  const int tid = threadIdx.x;
  const int lane = tid & 31, wave = tid >> 5;
  const int lm = lane & 15, khalf = lane >> 4;
  const int kblk = blockIdx.x;  // 0..7

  for (int i = tid; i < BB * 64; i += 256) {
    int b = i >> 6, j = i & 63;
    c_s[b][j] = c0[b * HID + kblk * 64 + j];
  }

  for (int t = 0; t < TT; ++t) {
    __syncthreads();
    // Snapshot masked h_prev into LDS; mask c locally.
    for (int i = tid; i < BB * HID; i += 256) {
      int b = i / HID, kk = i % HID;
      float mask = 1.f - done[t * BB + b];
      hp_s[b][kk] = (_Float16)((float)hbuf[i] * mask);
    }
    for (int i = tid; i < BB * 64; i += 256) {
      int b = i >> 6;
      c_s[b][i & 63] *= (1.f - done[t * BB + b]);
    }
    grid_barrier(&sync[2 * t]);  // everyone snapshotted before anyone overwrites hbuf

    // gates[32, 256] = hp(f16)[32,512] @ Whh_slice^T ; wave w covers q in [w*32,w*32+32)
    v8f acc[2][2] = {};
    for (int kc = 0; kc < HID; kc += 32) {
      v16h aF[2], bF[2];
#pragma unroll
      for (int mi = 0; mi < 2; ++mi)
        aF[mi] = frag16(&hp_s[mi * 16 + lm][0], kc + khalf * 8);
#pragma unroll
      for (int ni = 0; ni < 2; ++ni) {
        int q = wave * 32 + ni * 16 + lm;                 // local gate col 0..255
        int row = (q >> 6) * HID + kblk * 64 + (q & 63);  // global W_hh row
        bF[ni] = frag16(Whh + (size_t)row * HID, kc + khalf * 8);
      }
#pragma unroll
      for (int mi = 0; mi < 2; ++mi)
#pragma unroll
        for (int ni = 0; ni < 2; ++ni)
          acc[mi][ni] = __builtin_amdgcn_wmma_f32_16x16x32_f16(
              false, aF[mi], false, bF[ni], (short)0, acc[mi][ni], false, false);
    }
    __syncthreads();  // done reading hp_s; safe to overlay gates_s

    const int moff = (lane >> 4) * 8;
#pragma unroll
    for (int mi = 0; mi < 2; ++mi)
#pragma unroll
      for (int ni = 0; ni < 2; ++ni) {
        int q = wave * 32 + ni * 16 + lm;
#pragma unroll
        for (int r = 0; r < 8; ++r) {
          int b = mi * 16 + moff + r;
          int col = (q >> 6) * HID + kblk * 64 + (q & 63);
          gates_s[b][q] = acc[mi][ni][r] + XG[(size_t)(t * BB + b) * 2048 + col];
        }
      }
    __syncthreads();

    // Cell update for this block's 64 hidden columns.
    for (int i = tid; i < BB * 64; i += 256) {
      int b = i >> 6, j = i & 63;
      float gi = sigmf(gates_s[b][j]);
      float gf = sigmf(gates_s[b][64 + j]);
      float gg = tanhf(gates_s[b][128 + j]);
      float go = sigmf(gates_s[b][192 + j]);
      float c = gf * c_s[b][j] + gi * gg;
      float h = go * tanhf(c);
      c_s[b][j] = c;
      int col = kblk * 64 + j;
      feat[(size_t)(t * BB + b) * HID + col] = (_Float16)h;
      hbuf[b * HID + col] = (_Float16)h;
      if (t == TT - 1) {
        outHN[b * HID + col] = h;
        outCN[b * HID + col] = c;
      }
    }
    grid_barrier(&sync[2 * t + 1]);  // all slices of h_t written
  }
}

// ---------------------------------------------------------------------------
// Host-side orchestration
// ---------------------------------------------------------------------------
static inline char* bump(char*& p, size_t bytes) {
  char* r = p;
  p += (bytes + 255) & ~(size_t)255;
  return r;
}

extern "C" void kernel_launch(void* const* d_in, const int* in_sizes, int n_in,
                              void* d_out, int out_size, void* d_ws, size_t ws_size,
                              hipStream_t stream) {
  (void)in_sizes; (void)n_in; (void)out_size; (void)ws_size;
  const float* x       = (const float*)d_in[0];
  const float* done    = (const float*)d_in[1];
  const float* h0      = (const float*)d_in[2];
  const float* c0      = (const float*)d_in[3];
  const float* conv1_w = (const float*)d_in[4];
  const float* conv1_b = (const float*)d_in[5];
  const float* conv2_w = (const float*)d_in[6];
  const float* conv2_b = (const float*)d_in[7];
  const float* conv3_w = (const float*)d_in[8];
  const float* conv3_b = (const float*)d_in[9];
  const float* fc1_w   = (const float*)d_in[10];
  const float* fc1_b   = (const float*)d_in[11];
  const float* fc2_w   = (const float*)d_in[12];
  const float* fc2_b   = (const float*)d_in[13];
  const float* w_ih    = (const float*)d_in[14];
  const float* w_hh    = (const float*)d_in[15];
  const float* b_ih    = (const float*)d_in[16];
  const float* b_hh    = (const float*)d_in[17];
  const float* df_w    = (const float*)d_in[18];
  const float* df_b    = (const float*)d_in[19];
  const float* disc_w  = (const float*)d_in[20];
  const float* disc_b  = (const float*)d_in[21];
  float* out = (float*)d_out;

  char* w = (char*)d_ws;
  _Float16* A1      = (_Float16*)bump(w, (size_t)ROWS * FEAT_IN * 2);
  _Float16* fc1w16  = (_Float16*)bump(w, (size_t)HID * FEAT_IN * 2);
  _Float16* fc2w16  = (_Float16*)bump(w, (size_t)HID * HID * 2);
  _Float16* wih16   = (_Float16*)bump(w, (size_t)4 * HID * HID * 2);
  _Float16* whh16   = (_Float16*)bump(w, (size_t)4 * HID * HID * 2);
  _Float16* dfw16   = (_Float16*)bump(w, (size_t)128 * HID * 2);
  _Float16* discw16 = (_Float16*)bump(w, (size_t)ZD * 128 * 2);
  _Float16* H1      = (_Float16*)bump(w, (size_t)ROWS * HID * 2);
  _Float16* H2      = (_Float16*)bump(w, (size_t)ROWS * HID * 2);
  float*    XG      = (float*)bump(w, (size_t)ROWS * 4 * HID * 4);
  _Float16* featb   = (_Float16*)bump(w, (size_t)ROWS * HID * 2);
  _Float16* hbuf    = (_Float16*)bump(w, (size_t)BB * HID * 2);
  _Float16* D1      = (_Float16*)bump(w, (size_t)ROWS * 128 * 2);
  unsigned int* sync = (unsigned int*)bump(w, 4096);

  // 0) reset barrier counters (graph replays don't re-poison ws)
  k_zero_u32<<<1, 128, 0, stream>>>(sync, 128);

  // 1) weight / state conversions to f16
  auto cvt = [&](const float* s, _Float16* d, int n) {
    k_cvt_f32_f16<<<(n + 255) / 256, 256, 0, stream>>>(s, d, n);
  };
  cvt(fc1_w,  fc1w16,  HID * FEAT_IN);
  cvt(fc2_w,  fc2w16,  HID * HID);
  cvt(w_ih,   wih16,   4 * HID * HID);
  cvt(w_hh,   whh16,   4 * HID * HID);
  cvt(df_w,   dfw16,   128 * HID);
  cvt(disc_w, discw16, ZD * 128);
  cvt(h0,     hbuf,    BB * HID);

  // 2) fused convs -> f16 A matrix [2048, 28224]
  k_conv_fused<<<dim3(6, 6, ROWS), 256, 0, stream>>>(
      x, conv1_w, conv1_b, conv2_w, conv2_b, conv3_w, conv3_b, A1);

  // 3) FC1: H1 = lrelu(A1 @ fc1_w^T + b)   [2048,512], K=28224  (dominant GEMM)
  k_wmma_gemm<<<dim3(ROWS / 128, HID / 64), 256, 0, stream>>>(
      A1, fc1w16, fc1_b, nullptr, H1, nullptr, ROWS, HID, FEAT_IN, 1);

  // 4) FC2: H2 = lrelu(H1 @ fc2_w^T + b)   [2048,512], K=512
  k_wmma_gemm<<<dim3(ROWS / 128, HID / 64), 256, 0, stream>>>(
      H1, fc2w16, fc2_b, nullptr, H2, nullptr, ROWS, HID, HID, 1);

  // 5) XG = H2 @ w_ih^T + b_ih + b_hh      [2048,2048] f32, K=512
  k_wmma_gemm<<<dim3(ROWS / 128, (4 * HID) / 64), 256, 0, stream>>>(
      H2, wih16, b_ih, b_hh, nullptr, XG, ROWS, 4 * HID, HID, 0);

  // 6) sequential LSTM (8 cooperating blocks, WMMA recurrent GEMM)
  k_lstm_seq<<<dim3(LSTM_BLOCKS), 256, 0, stream>>>(
      XG, whh16, done, c0, hbuf, featb,
      out + ROWS * ZD, out + ROWS * ZD + BB * HID, sync);

  // 7) D1 = lrelu(feat @ df_w^T + b)       [2048,128], K=512
  k_wmma_gemm<<<dim3(ROWS / 128, 128 / 64), 256, 0, stream>>>(
      featb, dfw16, df_b, nullptr, D1, nullptr, ROWS, 128, HID, 1);

  // 8) logits = D1 @ disc_w^T + b -> d_out [2048,64] f32, K=128
  k_wmma_gemm<<<dim3(ROWS / 128, ZD / 64), 256, 0, stream>>>(
      D1, discw16, disc_b, nullptr, nullptr, out, ROWS, ZD, 128, 0);
}
